// SpatialFeatureEnhance_256_75505525064603
// MI455X (gfx1250) — matrix operand order repaired and checked by reading, not compiled, not run on hardware
//
#include <hip/hip_runtime.h>
#include <hip/hip_bf16.h>

typedef __attribute__((ext_vector_type(16))) __bf16 bf16x16;
typedef __attribute__((ext_vector_type(8)))  __bf16 bf16x8;
typedef __attribute__((ext_vector_type(8)))  float  f32x8;
typedef __attribute__((ext_vector_type(2)))  float  f32x2;

#define C_   256
#define H_   96
#define W_   96
#define WIN_ 7
#define PAD_ 3
#define TH_  4                 // output rows per workgroup
#define TW_  16                // output cols per workgroup
#define NPX  64                // TH_*TW_ output pixels per workgroup
#define APX  22                // staged halo width  (TW_ + 6)
#define ARR  10                // staged halo rows   (TH_ + 6)
#define AS   260               // LDS pixel stride in shorts (256 + 4 pad -> bank advance 2)

// ---------------------------------------------------------------------------
// Precompose: W0s = sc0_w @ rm_w [6x49], b0s = sc0_w @ rm_b + sc0_b [6]
//             W0h = sh0_w @ rm_w [6x49], b0h = sh0_w @ rm_b + sh0_b [6]
// ws floats: [0,294) W0s | [294,300) b0s | [300,594) W0h | [594,600) b0h
// ---------------------------------------------------------------------------
__global__ void sfe_precompose(const float* __restrict__ rm_w,
                               const float* __restrict__ rm_b,
                               const float* __restrict__ sc0_w,
                               const float* __restrict__ sc0_b,
                               const float* __restrict__ sh0_w,
                               const float* __restrict__ sh0_b,
                               float* __restrict__ ws) {
    int t = threadIdx.x;
    for (int idx = t; idx < 6 * 49 * 2; idx += blockDim.x) {
        int half = idx / (6 * 49);
        int r    = idx % (6 * 49);
        int f = r / 49, k = r % 49;
        const float* w0 = half ? sh0_w : sc0_w;
        float acc = 0.f;
        for (int c = 0; c < C_; ++c) acc += w0[f * C_ + c] * rm_w[c * 49 + k];
        ws[half * 300 + f * 49 + k] = acc;
    }
    for (int idx = t; idx < 12; idx += blockDim.x) {
        int half = idx / 6, f = idx % 6;
        const float* w0 = half ? sh0_w : sc0_w;
        const float* b0 = half ? sh0_b : sc0_b;
        float acc = b0[f];
        for (int c = 0; c < C_; ++c) acc += w0[f * C_ + c] * rm_b[c];
        ws[half * 300 + 294 + f] = acc;
    }
}

// ---------------------------------------------------------------------------
// Fused main kernel, one WG per (4 rows x 16 cols, batch).
//  stage(bf16,LDS) -> corr band-GEMM (v_wmma bf16) -> softmax+6x49 matvec ->
//  expand 6->256 via v_wmma f32 16x16x4 -> out = a1*(scale+1)+shift
// ---------------------------------------------------------------------------
__global__ void __launch_bounds__(512)
sfe_main(const float* __restrict__ a0, const float* __restrict__ a1,
         const float* __restrict__ ws,
         const float* __restrict__ sc1_w, const float* __restrict__ sc1_b,
         const float* __restrict__ sh1_w, const float* __restrict__ sh1_b,
         float* __restrict__ out) {
    __shared__ __align__(32) unsigned short s_a0[NPX * AS];        // 33.3 KB
    __shared__ __align__(32) unsigned short s_a1[ARR * APX * AS];  // 114.4 KB
    __shared__ __align__(32) float s_corr[NPX * 49];               // 12.5 KB
    __shared__ __align__(32) float s_us[NPX * 8];                  // f=6,7 zero pad
    __shared__ __align__(32) float s_uh[NPX * 8];

    const int w0   = blockIdx.x * TW_;
    const int h0   = blockIdx.y * TH_;
    const int b    = blockIdx.z;
    const int tid  = threadIdx.x;
    const int lane = tid & 31;
    const int wave = tid >> 5;
    const int n16  = lane & 15;
    const int khalf = lane >> 4;
    const size_t HW = (size_t)H_ * W_;

    // ================= stage a0: 64 px x 256 ch, bf16 packed pairs =========
    // idx -> wcol (coalesced), lr, even channel c2; v_perm packs 2 ch / dword
    #pragma unroll 4
    for (int it = 0; it < 16; ++it) {
        int idx  = tid + 512 * it;
        int wcol = idx & 15;
        int lr   = (idx >> 4) & 3;
        int c2   = (idx >> 6) * 2;
        const float* p = a0 + ((size_t)(b * C_ + c2) * H_ + (h0 + lr)) * W_ + w0 + wcol;
        unsigned u0 = __float_as_uint(p[0])  + 0x8000u;
        unsigned u1 = __float_as_uint(p[HW]) + 0x8000u;
        *(unsigned*)&s_a0[(lr * 16 + wcol) * AS + c2] =
            __builtin_amdgcn_perm(u1, u0, 0x07060302u);
    }
    // ================= stage a1 halo: 10 rows x 22 px x 256 ch ==============
    {
        const int px = lane;                      // 0..31, valid < 22
        const int wg = w0 - PAD_ + px;
        const bool wok = (unsigned)wg < (unsigned)W_;
        if (px < APX) {
            for (int rr = 0; rr < ARR; ++rr) {
                const int hh = h0 - PAD_ + rr;
                const bool ok = wok && ((unsigned)hh < (unsigned)H_);
                const long boff = (long)b * C_ * (long)HW + (long)hh * W_ + wg;
                const unsigned sb = (rr * APX + px) * AS;
                #pragma unroll
                for (int it = 0; it < 8; ++it) {
                    int c2 = (wave + 16 * it) * 2;
                    float f0 = ok ? a1[boff + (long)c2 * (long)HW] : 0.f;
                    float f1 = ok ? a1[boff + (long)(c2 + 1) * (long)HW] : 0.f;
                    unsigned u0 = __float_as_uint(f0) + 0x8000u;
                    unsigned u1 = __float_as_uint(f1) + 0x8000u;
                    *(unsigned*)&s_a1[sb + c2] =
                        __builtin_amdgcn_perm(u1, u0, 0x07060302u);
                }
            }
        }
    }
    __syncthreads();

    // ================= corr band-GEMM (bf16 WMMA) ===========================
    // Task t: wb = t&3 (4x4 spatial A block along w), rr = t>>2 (a1 halo row).
    // D[m,n] = <a0 px(m), a1[rr, w0+wb*4-3+n]> ; i = rr-(m>>2), dj = n-(m&3).
    {
        const int wb = wave & 3;
        // A fragment: pixel(m=n16) = (lr)*16 + wb*4 + lc ; 16-bit A interleave:
        // VGPR0-7 = K[khalf*16,+16) (contiguous, matches B)
        const unsigned short* abase =
            &s_a0[(((n16 >> 2) * 16) + wb * 4 + (n16 & 3)) * AS + khalf * 16];
        bf16x16 afr[8];
        #pragma unroll
        for (int kk = 0; kk < 8; ++kk) {
            bf16x8 lo = *(const bf16x8*)(const void*)(abase + kk * 32);
            bf16x8 hi = *(const bf16x8*)(const void*)(abase + kk * 32 + 8);
            afr[kk] = __builtin_shufflevector(lo, hi,
                        0, 1, 2, 3, 4, 5, 6, 7, 8, 9, 10, 11, 12, 13, 14, 15);
        }
        for (int j = 0; j < 3; ++j) {
            const int t = wave + 16 * j;     // t&3 == wb  (16 = 0 mod 4)
            if (t >= 40) break;              // wave-uniform
            const int rr = t >> 2;
            int bpx = wb * 4 + n16;          // band only uses n <= lc+6 -> px <= 21
            if (bpx > APX - 1) bpx = APX - 1;
            const unsigned short* bbase = &s_a1[(rr * APX + bpx) * AS + khalf * 16];
            f32x8 acc = {};
            #pragma unroll
            for (int kk = 0; kk < 8; ++kk) {
                bf16x16 bfr = *(const bf16x16*)(const void*)(bbase + kk * 32);
                acc = __builtin_amdgcn_wmma_f32_16x16x32_bf16(
                          false, afr[kk], false, bfr, (short)0, acc, false, false);
            }
            #pragma unroll
            for (int r = 0; r < 8; ++r) {
                int m  = r + 8 * khalf;
                int i  = rr - (m >> 2);
                int dj = n16 - (m & 3);
                if (((unsigned)i < WIN_) && ((unsigned)dj < WIN_)) {
                    int px64 = (m >> 2) * 16 + wb * 4 + (m & 3);
                    s_corr[px64 * 49 + i * WIN_ + dj] = acc[r] * 0.0625f; // 1/sqrt(256)
                }
            }
        }
    }
    __syncthreads();

    // ================= softmax(49) + fused 6x49 matvecs (64 threads) ========
    if (tid < NPX) {
        float cr[49];
        float mx = -3.4e38f;
        #pragma unroll
        for (int k = 0; k < 49; ++k) { cr[k] = s_corr[tid * 49 + k]; mx = fmaxf(mx, cr[k]); }
        float sum = 0.f;
        #pragma unroll
        for (int k = 0; k < 49; ++k) { cr[k] = __expf(cr[k] - mx); sum += cr[k]; }
        const float inv = 1.0f / sum;
        #pragma unroll
        for (int k = 0; k < 49; ++k) cr[k] *= inv;
        #pragma unroll
        for (int f = 0; f < 6; ++f) {
            float ts = ws[294 + f];
            float th = ws[594 + f];
            #pragma unroll
            for (int k = 0; k < 49; ++k) {
                ts += ws[f * 49 + k]       * cr[k];
                th += ws[300 + f * 49 + k] * cr[k];
            }
            s_us[tid * 8 + f] = (ts >= 0.f) ? ts : 0.1f * ts;   // lrelu
            s_uh[tid * 8 + f] = (th >= 0.f) ? th : 0.1f * th;
        }
        s_us[tid * 8 + 6] = 0.f; s_us[tid * 8 + 7] = 0.f;       // K pad
        s_uh[tid * 8 + 6] = 0.f; s_uh[tid * 8 + 7] = 0.f;
    }
    __syncthreads();

    // ================= 6->256 expansion via f32 WMMA 16x16x4 ================
    // D[c_row, w_col] : M = 16 channels (c0 = wave*16), N = 16 w-cols, K = 6(+2 pad).
    // f32 A 16x4: lane holds M=lane&15, K = {2*khalf, 2*khalf+1} in 2 VGPRs.
    // f32 B 4x16: lane holds N=lane&15, K = {2*khalf, 2*khalf+1}.
    {
        const int c0 = wave * 16;
        const int crow = c0 + n16;
        const f32x2 zero2 = {0.f, 0.f};
        const float* wrs = sc1_w + crow * 6;
        const float* wrh = sh1_w + crow * 6;
        f32x2 As0 = *(const f32x2*)(wrs + 2 * khalf);                       // K 0..3
        f32x2 Ah0 = *(const f32x2*)(wrh + 2 * khalf);
        f32x2 As1 = (khalf == 0) ? *(const f32x2*)(wrs + 4) : zero2;        // K 4,5 | 6,7=0
        f32x2 Ah1 = (khalf == 0) ? *(const f32x2*)(wrh + 4) : zero2;

        f32x8 Cs = *(const f32x8*)(sc1_b + c0 + 8 * khalf);                 // bias rows
        #pragma unroll
        for (int r = 0; r < 8; ++r) Cs[r] += 1.0f;                          // fold (scale+1)
        f32x8 Ch = *(const f32x8*)(sh1_b + c0 + 8 * khalf);

        #pragma unroll
        for (int lr = 0; lr < TH_; ++lr) {
            const float* ub  = s_us + (lr * 16 + n16) * 8;
            const float* uhb = s_uh + (lr * 16 + n16) * 8;
            f32x2 Bs0 = *(const f32x2*)(ub + 2 * khalf);
            f32x2 Bs1 = *(const f32x2*)(ub + 4 + 2 * khalf);                // f=6,7 are zeros
            f32x2 Bh0 = *(const f32x2*)(uhb + 2 * khalf);
            f32x2 Bh1 = *(const f32x2*)(uhb + 4 + 2 * khalf);

            f32x8 Dsc = __builtin_amdgcn_wmma_f32_16x16x4_f32(
                            false, As0, false, Bs0, (short)0, Cs, false, false);
            Dsc = __builtin_amdgcn_wmma_f32_16x16x4_f32(
                            false, As1, false, Bs1, (short)0, Dsc, false, false);
            f32x8 Dsh = __builtin_amdgcn_wmma_f32_16x16x4_f32(
                            false, Ah0, false, Bh0, (short)0, Ch, false, false);
            Dsh = __builtin_amdgcn_wmma_f32_16x16x4_f32(
                            false, Ah1, false, Bh1, (short)0, Dsh, false, false);

            const size_t gbase =
                ((size_t)(b * C_ + c0 + 8 * khalf) * H_ + (h0 + lr)) * W_ + w0 + n16;
            #pragma unroll
            for (int r = 0; r < 8; ++r) {       // rows c = c0 + 8*khalf + r
                size_t gi = gbase + (size_t)r * HW;
                out[gi] = a1[gi] * Dsc[r] + Dsh[r];   // Dsc already = scale+1
            }
        }
    }
}

extern "C" void kernel_launch(void* const* d_in, const int* in_sizes, int n_in,
                              void* d_out, int out_size, void* d_ws, size_t ws_size,
                              hipStream_t stream) {
    const float* a0    = (const float*)d_in[0];
    const float* a1    = (const float*)d_in[1];
    const float* rm_w  = (const float*)d_in[2];
    const float* rm_b  = (const float*)d_in[3];
    const float* sc0_w = (const float*)d_in[4];
    const float* sc0_b = (const float*)d_in[5];
    const float* sc1_w = (const float*)d_in[6];
    const float* sc1_b = (const float*)d_in[7];
    const float* sh0_w = (const float*)d_in[8];
    const float* sh0_b = (const float*)d_in[9];
    const float* sh1_w = (const float*)d_in[10];
    const float* sh1_b = (const float*)d_in[11];
    float* ws  = (float*)d_ws;
    float* out = (float*)d_out;

    hipLaunchKernelGGL(sfe_precompose, dim3(1), dim3(256), 0, stream,
                       rm_w, rm_b, sc0_w, sc0_b, sh0_w, sh0_b, ws);

    dim3 grid(W_ / TW_, H_ / TH_, 8);   // (6, 24, 8)
    hipLaunchKernelGGL(sfe_main, grid, dim3(512), 0, stream,
                       a0, a1, ws, sc1_w, sc1_b, sh1_w, sh1_b, out);
}
